// MixerBlock_33011118637635
// MI455X (gfx1250) — compile-verified, matrix operand order checked by reading, not run on hardware
//
#include <hip/hip_runtime.h>
#include <math.h>

// ---------------------------------------------------------------------------
// MixerBlock for MI455X (gfx1250): bf16 WMMA GEMMs + O(T) decay scan.
//   B=8, T=2048, E=1024, H=16, HD=64, DFF=4096, DC=4
// Pipeline:
//   castT(out_w/ff_w1/ff_w2) + cast(w_proj)  -> bf16 W^T
//   LN1(x) -> h_bf16
//   GEMM p = h_bf16 @ Wproj + b_proj                (f32 out)
//   scan: mixed = f(s)*cumsum_r(g(t)*p) + mix_b     (bf16 out)
//   GEMM x1 = mixed @ out_w + out_b + x             (f32 out)
//   LN2(x1) -> h2_bf16
//   GEMM g = gelu(h2 @ ff_w1 + ff_b1)               (bf16 out)
//   GEMM y = g @ ff_w2 + ff_b2 + x1                 (f32 out = d_out)
// Global->LDS staging uses gfx1250 async-to-LDS (ASYNCcnt) when available.
// Workspace: ~244 MB assumed available in d_ws.
// ---------------------------------------------------------------------------

typedef __attribute__((ext_vector_type(16))) __bf16 v16bf;
typedef __attribute__((ext_vector_type(8)))  float  v8f;
typedef __attribute__((ext_vector_type(4)))  int    v4i;

#define B_    8
#define T_    2048
#define E_    1024
#define H_    16
#define HD_   64
#define DFF_  4096
#define MROWS (B_ * T_)   // 16384

#if defined(__gfx1250__) && __has_builtin(__builtin_amdgcn_global_load_async_to_lds_b128)
#define USE_ASYNC_LDS 1
#else
#define USE_ASYNC_LDS 0
#endif

__device__ __forceinline__ unsigned short f2bf(float f) {
  __bf16 h = (__bf16)f;                       // RNE convert on gfx1250
  return __builtin_bit_cast(unsigned short, h);
}

__device__ __forceinline__ float gelu_tanh(float x) {
  const float c0 = 0.7978845608028654f;       // sqrt(2/pi)
  return 0.5f * x * (1.0f + tanhf(c0 * (x + 0.044715f * x * x * x)));
}

// 16 B/lane global -> LDS copy; async (no VGPR data path) when supported.
__device__ __forceinline__ void cp16(unsigned short* lds, const unsigned short* g) {
#if USE_ASYNC_LDS
  __builtin_amdgcn_global_load_async_to_lds_b128(
      (__attribute__((address_space(1))) v4i*)g,
      (__attribute__((address_space(3))) v4i*)lds,
      0 /*offset*/, 0 /*cpol*/);
#else
  *(uint4*)lds = *(const uint4*)g;
#endif
}

__device__ __forceinline__ void wait_cp() {
#if USE_ASYNC_LDS
#if __has_builtin(__builtin_amdgcn_s_wait_asynccnt)
  __builtin_amdgcn_s_wait_asynccnt(0);
#else
  asm volatile("s_wait_asynccnt 0x0" ::: "memory");
#endif
#endif
}

// ---------------------------------------------------------------------------
// LayerNorm over E=1024, one row per block (256 threads, float4 per thread),
// writes bf16 normalized activations.
// ---------------------------------------------------------------------------
__global__ __launch_bounds__(256) void ln_bf16_kernel(
    const float* __restrict__ x, const float* __restrict__ g,
    const float* __restrict__ b, unsigned short* __restrict__ out)
{
  const int row = blockIdx.x;
  const int tid = threadIdx.x;
  const float4 v = ((const float4*)(x + (size_t)row * E_))[tid];
  float s  = v.x + v.y + v.z + v.w;
  float s2 = v.x*v.x + v.y*v.y + v.z*v.z + v.w*v.w;
  #pragma unroll
  for (int o = 16; o > 0; o >>= 1) { s += __shfl_xor(s, o); s2 += __shfl_xor(s2, o); }
  __shared__ float ws[8], ws2[8];
  const int wave = tid >> 5;
  if ((tid & 31) == 0) { ws[wave] = s; ws2[wave] = s2; }
  __syncthreads();
  float tot = 0.f, tot2 = 0.f;
  #pragma unroll
  for (int i = 0; i < 8; ++i) { tot += ws[i]; tot2 += ws2[i]; }
  const float mu  = tot  * (1.0f / E_);
  const float var = tot2 * (1.0f / E_) - mu * mu;
  const float rs  = rsqrtf(var + 1e-5f);
  const float4 gv = ((const float4*)g)[tid];
  const float4 bv = ((const float4*)b)[tid];
  ushort4 o4;
  o4.x = f2bf((v.x - mu) * rs * gv.x + bv.x);
  o4.y = f2bf((v.y - mu) * rs * gv.y + bv.y);
  o4.z = f2bf((v.z - mu) * rs * gv.z + bv.z);
  o4.w = f2bf((v.w - mu) * rs * gv.w + bv.w);
  ((ushort4*)(out + (size_t)row * E_))[tid] = o4;
}

// ---------------------------------------------------------------------------
// Cast+transpose f32 W[K][N] -> bf16 W^T[N][K] (tiled, conflict-free).
// ---------------------------------------------------------------------------
__global__ __launch_bounds__(256) void castT_kernel(
    const float* __restrict__ W, unsigned short* __restrict__ Wt, int K, int N)
{
  __shared__ float tile[32][33];
  const int n0 = blockIdx.x * 32;
  const int k0 = blockIdx.y * 32;
  const int tx = threadIdx.x, ty = threadIdx.y;
  #pragma unroll
  for (int i = ty; i < 32; i += 8)
    tile[i][tx] = W[(size_t)(k0 + i) * N + n0 + tx];
  __syncthreads();
  #pragma unroll
  for (int i = ty; i < 32; i += 8)
    Wt[(size_t)(n0 + i) * K + k0 + tx] = f2bf(tile[tx][i]);
}

// w_proj (H,E,HD) -> bf16 W^T[n=h*64+f][e]  (combined head projection weight)
__global__ __launch_bounds__(256) void cast_wproj_kernel(
    const float* __restrict__ wp, unsigned short* __restrict__ Wt)
{
  const int o = blockIdx.x * 256 + threadIdx.x;  // 0 .. E*E-1, o = n*E + e
  const int n = o / E_;
  const int e = o % E_;
  const int h = n >> 6, f = n & 63;
  Wt[(size_t)o] = f2bf(wp[(size_t)h * E_ * HD_ + (size_t)e * HD_ + f]);
}

// ---------------------------------------------------------------------------
// Causal decay mixing as an O(T) scan (1-semiseparable matrix):
//   cum[s] = r*cum[s-1] + g(t)*p[s] ;  out[s] = f(s)*cum[s] + mix_b[h,s]
// Col heads (h<8):  f(s)=mix_w[h,s], g=1.  Row heads: g(t)=mix_w[h,t], f=1.
// One thread per (b, h*64+f); wave-coalesced along features.
// ---------------------------------------------------------------------------
__global__ __launch_bounds__(256) void mix_scan_kernel(
    const float* __restrict__ p,          // (B*T) x E f32 (head-major feats)
    const float* __restrict__ mix_w,      // H x T
    const float* __restrict__ mix_b,      // H x T
    const float* __restrict__ decay,      // H
    unsigned short* __restrict__ mixed)   // (B*T) x E bf16
{
  const int idx = blockIdx.x * blockDim.x + threadIdx.x;  // 0 .. B*E-1
  const int b  = idx / E_;
  const int hf = idx % E_;
  const int h  = hf >> 6;
  float d = decay[h];
  d = fminf(fmaxf(d, 0.9f), 1.0f);
  const float r = powf(d, 0.25f);                 // d^(1/DC), DC=4
  const bool rowHead = (h >= (H_ / 2));
  const float* mw = mix_w + (size_t)h * T_;
  const float* mb = mix_b + (size_t)h * T_;
  const float* pp = p     + (size_t)b * T_ * E_ + hf;
  unsigned short* op = mixed + (size_t)b * T_ * E_ + hf;
  float cum = 0.f;
  #pragma unroll 4
  for (int t = 0; t < T_; ++t) {
    const float w  = mw[t];
    const float pv = pp[(size_t)t * E_];
    cum = cum * r + (rowHead ? w : 1.0f) * pv;
    const float ov = (rowHead ? cum : w * cum) + mb[t];
    op[(size_t)t * E_] = f2bf(ov);
  }
}

// ---------------------------------------------------------------------------
// bf16 WMMA GEMM: C[M,N] = A[M,K] * W[K,N] (+bias / +resid / gelu->bf16).
// A row-major bf16; W supplied transposed: Wt[N][K] row-major bf16.
// Macro-tile 128x128x32, 8 waves (2x4), wave tile 64x32 = 4x2 WMMA accums.
// Double-buffered LDS, pitch 40 bf16 (conflict-free ds_load_b128 fragments).
// Staging via GLOBAL_LOAD_ASYNC_TO_LDS_B128 (ASYNCcnt), overlapped with WMMA.
// Fragment layout (wave32): lanes 0-15 row m / K{0..7,16..23},
//                           lanes 16-31 row m / K{8..15,24..31}.
// ---------------------------------------------------------------------------
#define BM 128
#define BN 128
#define BK 32
#define LPITCH 40

template <int EPI>   // 0: bias->f32 ; 1: bias+resid->f32 ; 2: bias+gelu->bf16
__global__ __launch_bounds__(256) void gemm_bf16_kernel(
    const unsigned short* __restrict__ A,
    const unsigned short* __restrict__ Wt,
    const float* __restrict__ bias,
    const float* __restrict__ resid,
    float* __restrict__ outF,
    unsigned short* __restrict__ outH,
    int M, int N, int K)
{
  __shared__ unsigned short As[2][BM * LPITCH];
  __shared__ unsigned short Bs[2][BN * LPITCH];

  const int tid  = threadIdx.x;
  const int lane = tid & 31;
  const int wav  = tid >> 5;
  const int wm   = wav >> 2;     // 0..1
  const int wn   = wav & 3;      // 0..3

  const int m0 = blockIdx.y * BM;
  const int n0 = blockIdx.x * BN;

  // global->LDS staging: 512 slots of 8 bf16 (16B); 2 slots per thread
  const int r0 = tid >> 2;
  const int c0 = (tid & 3) * 8;
  const int r1 = r0 + 64;        // slot tid+256
  const unsigned short* gA0 = A  + (size_t)(m0 + r0) * K + c0;
  const unsigned short* gA1 = A  + (size_t)(m0 + r1) * K + c0;
  const unsigned short* gB0 = Wt + (size_t)(n0 + r0) * K + c0;
  const unsigned short* gB1 = Wt + (size_t)(n0 + r1) * K + c0;
  const int sA0 = r0 * LPITCH + c0;
  const int sA1 = r1 * LPITCH + c0;

  const v8f zero = {0.f, 0.f, 0.f, 0.f, 0.f, 0.f, 0.f, 0.f};
  v8f acc[4][2];
  #pragma unroll
  for (int i = 0; i < 4; ++i)
    #pragma unroll
    for (int j = 0; j < 2; ++j) acc[i][j] = zero;

  const int nk = K / BK;

  // prologue: stage k-tile 0 into buffer 0 (async)
  cp16(&As[0][sA0], gA0);
  cp16(&As[0][sA1], gA1);
  cp16(&Bs[0][sA0], gB0);
  cp16(&Bs[0][sA1], gB1);
  wait_cp();
  __syncthreads();

  const int rrow = lane & 15;
  const int koff = (lane >> 4) * 8;

  union Frag { v16bf v; uint4 q[2]; };

  for (int kt = 0; kt < nk; ++kt) {
    const int cur = kt & 1;

    // kick off async staging of the next k-tile before doing any math
    if (kt + 1 < nk) {
      const int ko  = (kt + 1) * BK;
      const int nxt = cur ^ 1;
      cp16(&As[nxt][sA0], gA0 + ko);
      cp16(&As[nxt][sA1], gA1 + ko);
      cp16(&Bs[nxt][sA0], gB0 + ko);
      cp16(&Bs[nxt][sA1], gB1 + ko);
    }

    Frag af[4], bf[2];
    #pragma unroll
    for (int i = 0; i < 4; ++i) {
      const unsigned short* p = &As[cur][(wm * 64 + i * 16 + rrow) * LPITCH + koff];
      af[i].q[0] = *(const uint4*)p;
      af[i].q[1] = *(const uint4*)(p + 16);
    }
    #pragma unroll
    for (int j = 0; j < 2; ++j) {
      const unsigned short* p = &Bs[cur][(wn * 32 + j * 16 + rrow) * LPITCH + koff];
      bf[j].q[0] = *(const uint4*)p;
      bf[j].q[1] = *(const uint4*)(p + 16);
    }

    #pragma unroll
    for (int i = 0; i < 4; ++i)
      #pragma unroll
      for (int j = 0; j < 2; ++j)
        acc[i][j] = __builtin_amdgcn_wmma_f32_16x16x32_bf16(
            false, af[i].v, false, bf[j].v, (short)0, acc[i][j], false, false);

    if (kt + 1 < nk) {
      wait_cp();          // async tile landed in LDS
      __syncthreads();    // visible to all waves before next compute
    }
  }

  // epilogue: C/D layout -> lanes 0-15 rows m..m+7 (VGPR r), lanes 16-31 rows m+8..m+15
  const int col  = lane & 15;
  const int rsel = (lane >> 4) * 8;
  #pragma unroll
  for (int j = 0; j < 2; ++j) {
    const int n = n0 + wn * 32 + j * 16 + col;
    const float bv = bias[n];
    #pragma unroll
    for (int i = 0; i < 4; ++i) {
      const int mb = m0 + wm * 64 + i * 16 + rsel;
      #pragma unroll
      for (int r = 0; r < 8; ++r) {
        const size_t idx = (size_t)(mb + r) * N + n;
        float v = acc[i][j][r] + bv;
        if (EPI == 1) v += resid[idx];
        if (EPI == 2) outH[idx] = f2bf(gelu_tanh(v));
        else          outF[idx] = v;
      }
    }
  }
}

// ---------------------------------------------------------------------------
extern "C" void kernel_launch(void* const* d_in, const int* in_sizes, int n_in,
                              void* d_out, int out_size, void* d_ws, size_t ws_size,
                              hipStream_t stream)
{
  const float* x      = (const float*)d_in[0];
  const float* w_proj = (const float*)d_in[1];
  const float* b_proj = (const float*)d_in[2];
  const float* mix_w  = (const float*)d_in[3];
  const float* mix_b  = (const float*)d_in[4];
  const float* decay  = (const float*)d_in[5];
  const float* out_w  = (const float*)d_in[6];
  const float* out_b  = (const float*)d_in[7];
  const float* ln1_g  = (const float*)d_in[8];
  const float* ln1_b  = (const float*)d_in[9];
  const float* ln2_g  = (const float*)d_in[10];
  const float* ln2_b  = (const float*)d_in[11];
  const float* ff_w1  = (const float*)d_in[12];
  const float* ff_b1  = (const float*)d_in[13];
  const float* ff_w2  = (const float*)d_in[14];
  const float* ff_b2  = (const float*)d_in[15];
  (void)in_sizes; (void)n_in; (void)out_size; (void)ws_size;

  char* ws = (char*)d_ws;
  size_t o = 0;
  unsigned short* wt_proj = (unsigned short*)(ws + o); o += (size_t)E_ * E_ * 2;
  unsigned short* wt_out  = (unsigned short*)(ws + o); o += (size_t)E_ * E_ * 2;
  unsigned short* wt_ff1  = (unsigned short*)(ws + o); o += (size_t)E_ * DFF_ * 2;
  unsigned short* wt_ff2  = (unsigned short*)(ws + o); o += (size_t)E_ * DFF_ * 2;
  // Region A (128 MB): first holds p (f32, 64 MB) + mixed (bf16, 32 MB);
  // both are dead before FFN1, so the whole region is reused for g (bf16).
  char* regA = ws + o; o += (size_t)MROWS * DFF_ * 2;
  float*          p_f32    = (float*)regA;
  unsigned short* mixed_bf = (unsigned short*)(regA + (size_t)MROWS * E_ * 4);
  unsigned short* g_bf     = (unsigned short*)regA;
  unsigned short* h_bf     = (unsigned short*)(ws + o); o += (size_t)MROWS * E_ * 2; // reused as h2
  float*          x1       = (float*)(ws + o);          o += (size_t)MROWS * E_ * 4;
  // total ~244 MB

  float* y = (float*)d_out;

  // 1) weight casts (bf16, transposed)
  castT_kernel<<<dim3(E_ / 32, E_ / 32), dim3(32, 8), 0, stream>>>(out_w, wt_out, E_, E_);
  castT_kernel<<<dim3(DFF_ / 32, E_ / 32), dim3(32, 8), 0, stream>>>(ff_w1, wt_ff1, E_, DFF_);
  castT_kernel<<<dim3(E_ / 32, DFF_ / 32), dim3(32, 8), 0, stream>>>(ff_w2, wt_ff2, DFF_, E_);
  cast_wproj_kernel<<<(E_ * E_) / 256, 256, 0, stream>>>(w_proj, wt_proj);

  // 2) LN1
  ln_bf16_kernel<<<MROWS, 256, 0, stream>>>(x, ln1_g, ln1_b, h_bf);

  // 3) head projection: p = h @ Wproj + b_proj   (b_proj flattens to bias[n], n=h*64+f)
  gemm_bf16_kernel<0><<<dim3(E_ / BN, MROWS / BM), 256, 0, stream>>>(
      h_bf, wt_proj, b_proj, nullptr, p_f32, nullptr, MROWS, E_, E_);

  // 4) causal decay mixing as linear scan (+ mix_b), bf16 out
  mix_scan_kernel<<<(B_ * E_) / 256, 256, 0, stream>>>(p_f32, mix_w, mix_b, decay, mixed_bf);

  // 5) out projection + residual: x1 = mixed @ out_w + out_b + x
  gemm_bf16_kernel<1><<<dim3(E_ / BN, MROWS / BM), 256, 0, stream>>>(
      mixed_bf, wt_out, out_b, x, x1, nullptr, MROWS, E_, E_);

  // 6) LN2
  ln_bf16_kernel<<<MROWS, 256, 0, stream>>>(x1, ln2_g, ln2_b, h_bf);

  // 7) FFN1 + gelu (bf16 out)
  gemm_bf16_kernel<2><<<dim3(DFF_ / BN, MROWS / BM), 256, 0, stream>>>(
      h_bf, wt_ff1, ff_b1, nullptr, nullptr, g_bf, MROWS, DFF_, E_);

  // 8) FFN2 + residual -> final output
  gemm_bf16_kernel<1><<<dim3(E_ / BN, MROWS / BM), 256, 0, stream>>>(
      g_bf, wt_ff2, ff_b2, x1, y, nullptr, MROWS, E_, DFF_);
}
